// gru_75814762709149
// MI455X (gfx1250) — matrix operand order repaired and checked by reading, not compiled, not run on hardware
//
#include <hip/hip_runtime.h>
#include <math.h>

// ---------------------------------------------------------------------------
// Types matching the gfx1250 WMMA builtin signatures
// ---------------------------------------------------------------------------
typedef __attribute__((ext_vector_type(16))) __bf16 v16bf;
typedef __attribute__((ext_vector_type(8)))  __bf16 v8bf;
typedef __attribute__((ext_vector_type(8)))  float  v8f;

// Workgroup = 256 threads = 8 waves (wave32), arranged 2 (M) x 4 (N).
// Workgroup tile: 128 (M) x 256 (N).  Wave tile 64x64 = 4x4 fragments of
// 16x16; K-step 32 -> 16 v_wmma_f32_16x16x32_bf16 per wave per step.
// LDS double-buffered (selected by OFFSET arithmetic, not pointer selects,
// so clang keeps addrspace(3) and emits ds_load/ds_store, not flat ops).
#define TM 128
#define TN 256
#define TK 32
#define LDS_STRIDE 40   // halfs per row (32 data + 8 pad), keeps 16B alignment
#define AS_BUF (TM * LDS_STRIDE)
#define BS_BUF (TN * LDS_STRIDE)

// ---------------------------------------------------------------------------
// Shared WMMA GEMM core: 128x256 workgroup tile of C = A[rows,K] @ Bt[cols,K]^T
// A row-major [numRows, K] bf16; Bt row-major [numCols, K] bf16 (B transposed)
// ---------------------------------------------------------------------------
__device__ __forceinline__ void gemm_tile_core(
    const __bf16* __restrict__ A,
    const __bf16* __restrict__ Bt,
    int K, int rowBase, int colBase,
    __bf16* AsBuf /*[2][AS_BUF]*/,
    __bf16* BsBuf /*[2][BS_BUF]*/,
    v8f c[4][4])
{
  const int tid   = threadIdx.x;
  const int lane  = tid & 31;
  const int wave  = tid >> 5;
  const int waveM = wave >> 2;        // 0..1 -> 64-row band
  const int waveN = wave & 3;         // 0..3 -> 64-col band
  const int r     = lane & 15;
  const int h     = lane >> 4;

  // staging split across 256 threads
  const int aRow  = tid >> 1;          // 0..127
  const int aCol  = (tid & 1) * 16;    // 0 or 16
  const int bRow0 = tid >> 1;          // 0..127
  const int bRow1 = bRow0 + 128;       // 128..255
  const int bCol  = (tid & 1) * 16;

  const __bf16* aSrcBase  = A  + (size_t)(rowBase + aRow)  * K + aCol;
  const __bf16* bSrc0Base = Bt + (size_t)(colBase + bRow0) * K + bCol;
  const __bf16* bSrc1Base = Bt + (size_t)(colBase + bRow1) * K + bCol;

  v8f zero = {};
  for (int i = 0; i < 4; ++i)
    for (int j = 0; j < 4; ++j)
      c[i][j] = zero;

  v8bf a0, a1, b0, b1, b2, b3;
  auto loadGlobal = [&](int k0) {
    const __bf16* as = aSrcBase + k0;
    a0 = *(const v8bf*)(as);
    a1 = *(const v8bf*)(as + 8);
    const __bf16* bs0 = bSrc0Base + k0;
    const __bf16* bs1 = bSrc1Base + k0;
    b0 = *(const v8bf*)(bs0);
    b1 = *(const v8bf*)(bs0 + 8);
    b2 = *(const v8bf*)(bs1);
    b3 = *(const v8bf*)(bs1 + 8);
    if (k0 + 2 * TK < K) {                       // global_prefetch_b8 (gfx1250)
      __builtin_prefetch(as + 2 * TK, 0, 1);
      __builtin_prefetch(bs0 + 2 * TK, 0, 1);
      __builtin_prefetch(bs1 + 2 * TK, 0, 1);
    }
  };
  // buf selects ping/pong by integer offset (keeps LDS addrspace inference)
  auto storeLds = [&](int buf) {
    __bf16* As = AsBuf + buf * AS_BUF;
    __bf16* Bs = BsBuf + buf * BS_BUF;
    *(v8bf*)(As + aRow * LDS_STRIDE + aCol)      = a0;
    *(v8bf*)(As + aRow * LDS_STRIDE + aCol + 8)  = a1;
    *(v8bf*)(Bs + bRow0 * LDS_STRIDE + bCol)     = b0;
    *(v8bf*)(Bs + bRow0 * LDS_STRIDE + bCol + 8) = b1;
    *(v8bf*)(Bs + bRow1 * LDS_STRIDE + bCol)     = b2;
    *(v8bf*)(Bs + bRow1 * LDS_STRIDE + bCol + 8) = b3;
  };
  auto compute = [&](int buf) {
    const __bf16* As = AsBuf + buf * AS_BUF;
    const __bf16* Bs = BsBuf + buf * BS_BUF;
    // B fragments: lane half selects K 0..15 / 16..31, column = lane&15
    v16bf bfrag[4];
    for (int j = 0; j < 4; ++j) {
      const __bf16* bp = Bs + (waveN * 64 + j * 16 + r) * LDS_STRIDE + h * 16;
      v8bf lo = *(const v8bf*)(bp);
      v8bf hi = *(const v8bf*)(bp + 8);
      for (int e = 0; e < 8; ++e) { bfrag[j][e] = lo[e]; bfrag[j][8 + e] = hi[e]; }
    }
    // A fragments: elems 0..15 = K[h*16..h*16+15]
    for (int i = 0; i < 4; ++i) {
      const __bf16* ap = As + (waveM * 64 + i * 16 + r) * LDS_STRIDE;
      v8bf lo = *(const v8bf*)(ap + h * 16);
      v8bf hi = *(const v8bf*)(ap + h * 16 + 8);
      v16bf afrag;
      for (int e = 0; e < 8; ++e) { afrag[e] = lo[e]; afrag[8 + e] = hi[e]; }
      for (int j = 0; j < 4; ++j) {
        c[i][j] = __builtin_amdgcn_wmma_f32_16x16x32_bf16(
            false, afrag, false, bfrag[j], (short)0, c[i][j], false, false);
      }
    }
  };

  const int nIter = K / TK;
  loadGlobal(0);
  storeLds(0);
  __syncthreads();
  for (int it = 0; it < nIter; ++it) {
    const int cur = it & 1;
    if (it + 1 < nIter) loadGlobal((it + 1) * TK);   // overlap with compute
    compute(cur);
    if (it + 1 < nIter) {
      storeLds(cur ^ 1);
      __syncthreads();
    }
  }
}

// ---------------------------------------------------------------------------
// GEMM with bias / accumulate epilogue:
//   Cout = A@B + (Cin ? Cin : 0) + (b1 ? b1[col] : 0) + (b2 ? b2[col] : 0)
// ---------------------------------------------------------------------------
__global__ __launch_bounds__(256) void gemm_bias_kernel(
    const __bf16* __restrict__ A, const __bf16* __restrict__ Bt,
    const float* __restrict__ Cin, const float* __restrict__ b1,
    const float* __restrict__ b2, float* __restrict__ Cout, int K, int ldc)
{
  __shared__ __align__(16) __bf16 AsBuf[2 * AS_BUF];
  __shared__ __align__(16) __bf16 BsBuf[2 * BS_BUF];
  const int rowBase = blockIdx.y * TM;
  const int colBase = blockIdx.x * TN;
  v8f c[4][4];
  gemm_tile_core(A, Bt, K, rowBase, colBase, AsBuf, BsBuf, c);

  const int lane  = threadIdx.x & 31;
  const int wave  = threadIdx.x >> 5;
  const int waveM = wave >> 2, waveN = wave & 3;
  const int r = lane & 15, h = lane >> 4;
  for (int i = 0; i < 4; ++i) {
    for (int j = 0; j < 4; ++j) {
      const int col = colBase + waveN * 64 + j * 16 + r;
      float bv = (b1 ? b1[col] : 0.f) + (b2 ? b2[col] : 0.f);
      for (int e = 0; e < 8; ++e) {
        const int row = rowBase + waveM * 64 + i * 16 + h * 8 + e;
        const size_t idx = (size_t)row * ldc + col;
        float v = c[i][j][e] + bv;
        if (Cin) v += Cin[idx];
        Cout[idx] = v;
      }
    }
  }
}

// ---------------------------------------------------------------------------
// Final GEMM with fused GRU epilogue:
//   h = tanh(G + A@B);  out = (1-z)*x + z*h
// ---------------------------------------------------------------------------
__global__ __launch_bounds__(256) void gemm_final_kernel(
    const __bf16* __restrict__ A, const __bf16* __restrict__ Bt,
    const float* __restrict__ G, const float* __restrict__ Z,
    const float* __restrict__ X, float* __restrict__ Out, int K, int ldc)
{
  __shared__ __align__(16) __bf16 AsBuf[2 * AS_BUF];
  __shared__ __align__(16) __bf16 BsBuf[2 * BS_BUF];
  const int rowBase = blockIdx.y * TM;
  const int colBase = blockIdx.x * TN;
  v8f c[4][4];
  gemm_tile_core(A, Bt, K, rowBase, colBase, AsBuf, BsBuf, c);

  const int lane  = threadIdx.x & 31;
  const int wave  = threadIdx.x >> 5;
  const int waveM = wave >> 2, waveN = wave & 3;
  const int r = lane & 15, h = lane >> 4;
  for (int i = 0; i < 4; ++i) {
    for (int j = 0; j < 4; ++j) {
      const int col = colBase + waveN * 64 + j * 16 + r;
      for (int e = 0; e < 8; ++e) {
        const int row = rowBase + waveM * 64 + i * 16 + h * 8 + e;
        const size_t idx = (size_t)row * ldc + col;
        const float hh = tanhf(G[idx] + c[i][j][e]);
        const float zz = Z[idx];
        Out[idx] = (1.f - zz) * X[idx] + zz * hh;
      }
    }
  }
}

// ---------------------------------------------------------------------------
// Elementwise helpers
// ---------------------------------------------------------------------------
__global__ void cvt_f32_bf16_kernel(const float* __restrict__ in,
                                    __bf16* __restrict__ out, size_t n)
{
  size_t i = (size_t)blockIdx.x * blockDim.x + threadIdx.x;
  if (i < n) out[i] = (__bf16)in[i];
}

// WT[m*K + k] = bf16(W[k*M + m])   (coalesced via LDS tile)
__global__ void transpose_cvt_kernel(const float* __restrict__ W,
                                     __bf16* __restrict__ WT, int Kd, int Md)
{
  __shared__ float tile[32][33];
  const int m0 = blockIdx.x * 32;
  const int k0 = blockIdx.y * 32;
  const int tx = threadIdx.x;   // 0..31
  const int ty = threadIdx.y;   // 0..7
  for (int yy = ty; yy < 32; yy += 8)
    tile[yy][tx] = W[(size_t)(k0 + yy) * Md + (m0 + tx)];
  __syncthreads();
  for (int yy = ty; yy < 32; yy += 8)
    WT[(size_t)(m0 + yy) * Kd + (k0 + tx)] = (__bf16)tile[tx][yy];
}

__global__ void rx_kernel(const float* __restrict__ R, const float* __restrict__ X,
                          __bf16* __restrict__ out, size_t n)
{
  size_t i = (size_t)blockIdx.x * blockDim.x + threadIdx.x;
  if (i < n) out[i] = (__bf16)(R[i] * X[i]);
}

// ---------------------------------------------------------------------------
// Host launcher
// ---------------------------------------------------------------------------
extern "C" void kernel_launch(void* const* d_in, const int* in_sizes, int n_in,
                              void* d_out, int out_size, void* d_ws, size_t ws_size,
                              hipStream_t stream)
{
  const int N = 8192, U = 2048, K = 2048;
  const size_t NU = (size_t)N * U;
  const size_t UU = (size_t)U * U;

  const float* x   = (const float*)d_in[0];
  const float* y   = (const float*)d_in[1];
  const float* Wz  = (const float*)d_in[2];
  const float* bz  = (const float*)d_in[3];
  const float* Uz  = (const float*)d_in[4];
  const float* buz = (const float*)d_in[5];
  const float* Wr  = (const float*)d_in[6];
  const float* br  = (const float*)d_in[7];
  const float* Ur  = (const float*)d_in[8];
  const float* bur = (const float*)d_in[9];
  const float* Wg  = (const float*)d_in[10];
  const float* bg  = (const float*)d_in[11];
  const float* Ug  = (const float*)d_in[12];
  const float* bug = (const float*)d_in[13];
  float* out = (float*)d_out;

  char* ws = (char*)d_ws;
  size_t off = 0;
  auto carve = [&](size_t bytes) { char* p = ws + off; off += bytes; return p; };

  __bf16* xbf  = (__bf16*)carve(NU * 2);
  __bf16* ybf  = (__bf16*)carve(NU * 2);
  __bf16* WzT  = (__bf16*)carve(UU * 2);
  __bf16* UzT  = (__bf16*)carve(UU * 2);
  __bf16* WrT  = (__bf16*)carve(UU * 2);
  __bf16* UrT  = (__bf16*)carve(UU * 2);
  __bf16* WgT  = (__bf16*)carve(UU * 2);
  __bf16* UgT  = (__bf16*)carve(UU * 2);
  float*  Zb   = (float*)carve(NU * 4);
  float*  Rb   = (float*)carve(NU * 4);
  float*  Gb   = (float*)carve(NU * 4);
  __bf16* rxbf = (__bf16*)carve(NU * 2);

  // 1) precision conversion of activations
  cvt_f32_bf16_kernel<<<(unsigned)(NU / 256), 256, 0, stream>>>(x, xbf, NU);
  cvt_f32_bf16_kernel<<<(unsigned)(NU / 256), 256, 0, stream>>>(y, ybf, NU);

  // 2) weight transpose + convert: WT[m][k] = W[k][m]
  dim3 tb(32, 8), tg(U / 32, K / 32);
  transpose_cvt_kernel<<<tg, tb, 0, stream>>>(Wz, WzT, K, U);
  transpose_cvt_kernel<<<tg, tb, 0, stream>>>(Uz, UzT, K, U);
  transpose_cvt_kernel<<<tg, tb, 0, stream>>>(Wr, WrT, K, U);
  transpose_cvt_kernel<<<tg, tb, 0, stream>>>(Ur, UrT, K, U);
  transpose_cvt_kernel<<<tg, tb, 0, stream>>>(Wg, WgT, K, U);
  transpose_cvt_kernel<<<tg, tb, 0, stream>>>(Ug, UgT, K, U);

  // 3) WMMA GEMMs
  dim3 gb(256), gg(U / TN, N / TM);
  // Z = y@Wz + bz + buz ; Z += x@Uz
  gemm_bias_kernel<<<gg, gb, 0, stream>>>(ybf, WzT, nullptr, bz, buz, Zb, K, U);
  gemm_bias_kernel<<<gg, gb, 0, stream>>>(xbf, UzT, Zb, nullptr, nullptr, Zb, K, U);
  // R = y@Wr + br + bur ; R += x@Ur
  gemm_bias_kernel<<<gg, gb, 0, stream>>>(ybf, WrT, nullptr, br, bur, Rb, K, U);
  gemm_bias_kernel<<<gg, gb, 0, stream>>>(xbf, UrT, Rb, nullptr, nullptr, Rb, K, U);
  // Gpre = y@Wg + bg + bug
  gemm_bias_kernel<<<gg, gb, 0, stream>>>(ybf, WgT, nullptr, bg, bug, Gb, K, U);

  // 4) rx = bf16(r * x)
  rx_kernel<<<(unsigned)(NU / 256), 256, 0, stream>>>(Rb, x, rxbf, NU);

  // 5) h = tanh(Gpre + rx@Ug); out = (1-z)*x + z*h
  gemm_final_kernel<<<gg, gb, 0, stream>>>(rxbf, UgT, Gb, Zb, x, out, K, U);
}
